// MultiHeadAttention_68247030334372
// MI455X (gfx1250) — matrix operand order repaired and checked by reading, not compiled, not run on hardware
//
#include <hip/hip_runtime.h>
#include <hip/hip_bf16.h>

typedef _Float16 half_t;
typedef __attribute__((ext_vector_type(16))) _Float16 v16h;
typedef __attribute__((ext_vector_type(8)))  _Float16 v8h;
typedef __attribute__((ext_vector_type(8)))  float    v8f;

#define T_DIM 2048
#define B_DIM 4
#define E_DIM 1024
#define H_DIM 16
#define DH    64
#define M_ROWS (T_DIM * B_DIM)   // 8192
#define N_QKV  (3 * E_DIM)       // 3072

// ---------- elementwise f32 -> f16 ----------
__global__ void k_cvt(const float* __restrict__ src, half_t* __restrict__ dst, int n) {
    int i = blockIdx.x * blockDim.x + threadIdx.x;
    if (i < n) dst[i] = (half_t)src[i];
}

// ---------- transpose + convert: src[rows][cols] f32 -> dst[cols][rows] f16 ----------
__global__ void k_tcvt(const float* __restrict__ src, half_t* __restrict__ dst,
                       int rows, int cols) {
    int i = blockIdx.x * blockDim.x + threadIdx.x;
    if (i < rows * cols) {
        int r = i / cols, c = i - r * cols;
        dst[(size_t)c * rows + r] = (half_t)src[i];
    }
}

// ---------- WMMA tile loads (wave32, 16-bit layouts per CDNA5 ISA 7.12.2) ----------
// A 16x32 (MxK): lane L holds row M=(L&15); K chunks {kb..kb+7} and {kb+16..kb+23}, kb=(L>>4)*8
__device__ __forceinline__ v16h load_A16x32(const half_t* __restrict__ base, int ld,
                                            int m0, int k0) {
    int lane = threadIdx.x & 31;
    int m  = m0 + (lane & 15);
    int kb = (lane >> 4) << 3;
    const half_t* p = base + (size_t)m * ld + k0 + kb;
    v8h lo = *(const v8h*)p;
    v8h hi = *(const v8h*)(p + 16);
    v16h a;
#pragma unroll
    for (int i = 0; i < 8; ++i) { a[i] = lo[i]; a[8 + i] = hi[i]; }
    return a;
}

// B 32x16 (KxN): lane L holds column N=(L&15); K chunks {kb..kb+7} and {kb+16..kb+23}, kb=(L>>4)*8
__device__ __forceinline__ v16h load_B32x16(const half_t* __restrict__ base, int ld,
                                            int k0, int n0) {
    int lane = threadIdx.x & 31;
    int kb = (lane >> 4) << 3;
    const half_t* p = base + (size_t)(k0 + kb) * ld + n0 + (lane & 15);
    v16h b;
#pragma unroll
    for (int i = 0; i < 8; ++i) { b[i] = p[(size_t)i * ld]; b[8 + i] = p[(size_t)(16 + i) * ld]; }
    return b;
}

__device__ __forceinline__ v8f wmma_f16(v16h a, v16h b, v8f c) {
    return __builtin_amdgcn_wmma_f32_16x16x32_f16(false, a, false, b, (short)0, c, false, false);
}

// ---------- GEMM1: qkv = x @ in_proj_w^T + b ; 32x64 register-blocked per wave ----------
__global__ void __launch_bounds__(32)
k_gemm_qkv(const half_t* __restrict__ X,      // [8192,1024] row-major f16
           const half_t* __restrict__ Wt,     // [1024,3072] (K-major) f16
           const float*  __restrict__ bias,   // [3072]
           half_t* __restrict__ qh,           // [B,H,T,DH] (q * Dh^-0.5)
           half_t* __restrict__ kT,           // [B,H,DH,T]
           half_t* __restrict__ vh) {         // [B,H,T,DH]
    int m0 = blockIdx.x * 32;
    int n0 = blockIdx.y * 64;
    v8f acc[2][4] = {};
    for (int k0 = 0; k0 < E_DIM; k0 += 32) {
        v16h a0 = load_A16x32(X, E_DIM, m0, k0);
        v16h a1 = load_A16x32(X, E_DIM, m0 + 16, k0);
#pragma unroll
        for (int j = 0; j < 4; ++j) {
            v16h b = load_B32x16(Wt, N_QKV, k0, n0 + 16 * j);
            acc[0][j] = wmma_f16(a0, b, acc[0][j]);
            acc[1][j] = wmma_f16(a1, b, acc[1][j]);
        }
    }
    int lane = threadIdx.x & 31;
#pragma unroll
    for (int i = 0; i < 2; ++i) {
#pragma unroll
        for (int j = 0; j < 4; ++j) {
            int n = n0 + 16 * j + (lane & 15);     // output feature f
            float bv = bias[n];
            int which = n >> 10;                   // 0=q 1=k 2=v
            int e = n & 1023;
            int h = e >> 6, d = e & 63;
            int mbase = m0 + 16 * i + ((lane >> 4) << 3);
#pragma unroll
            for (int r = 0; r < 8; ++r) {
                int m = mbase + r;
                int t = m >> 2, bb = m & 3;        // x rows are [t,b]
                int bh = bb * H_DIM + h;
                float val = acc[i][j][r] + bv;
                if (which == 0)      qh[((size_t)bh * T_DIM + t) * DH + d] = (half_t)(val * 0.125f);
                else if (which == 1) kT[((size_t)bh * DH + d) * T_DIM + t] = (half_t)val;
                else                 vh[((size_t)bh * T_DIM + t) * DH + d] = (half_t)val;
            }
        }
    }
}

// ---------- fused attention: scores -> softmax -> avg weights -> P@V ----------
__global__ void __launch_bounds__(128)
k_attn(const half_t* __restrict__ qh, const half_t* __restrict__ kT,
       const half_t* __restrict__ vh,
       half_t* __restrict__ attnh,    // [T,B,E] f16
       float*  __restrict__ avg) {    // [B,T,T] f32
    extern __shared__ char smemraw[];
    float  (*sc)[T_DIM]   = (float (*)[T_DIM])smemraw;                         // 128 KB f32 probs
    half_t (*pf)[T_DIM]   = (half_t(*)[T_DIM])(smemraw + 16 * T_DIM * 4);      //  64 KB f16 probs
    float  (*red)[8]      = (float (*)[8])(smemraw + 16 * T_DIM * 4 + 16 * T_DIM * 2);

    int b    = blockIdx.y;
    int t0   = blockIdx.x * 16;
    int tid  = threadIdx.x;
    int wave = tid >> 5;
    int lane = tid & 31;

    for (int h = 0; h < H_DIM; ++h) {
        int bh = b * H_DIM + h;
        const half_t* Q = qh + (size_t)bh * T_DIM * DH;
        const half_t* K = kT + (size_t)bh * DH * T_DIM;
        const half_t* V = vh + (size_t)bh * T_DIM * DH;

        // ---- phase A: S = Q K^T  (Q fragments hoisted; waves split s-tiles) ----
        {
            v16h qa0 = load_A16x32(Q, DH, t0, 0);
            v16h qa1 = load_A16x32(Q, DH, t0, 32);
            for (int st = wave; st < T_DIM / 16; st += 4) {
                int s0 = st * 16;
                v8f c = {};
                c = wmma_f16(qa0, load_B32x16(K, T_DIM, 0,  s0), c);
                c = wmma_f16(qa1, load_B32x16(K, T_DIM, 32, s0), c);
                int n = lane & 15;
                int rbase = (lane >> 4) << 3;
#pragma unroll
                for (int r = 0; r < 8; ++r) sc[rbase + r][s0 + n] = c[r];
            }
        }
        __syncthreads();

        // ---- phase B: softmax rows (8 threads per row); also emit f16 copy ----
        {
            int row = tid >> 3;
            int seg = tid & 7;
            int sbeg = seg * (T_DIM / 8);
            float mx = -3.4e38f;
            for (int s = sbeg; s < sbeg + T_DIM / 8; ++s) mx = fmaxf(mx, sc[row][s]);
            red[row][seg] = mx;
            __syncthreads();
            mx = red[row][0];
#pragma unroll
            for (int i = 1; i < 8; ++i) mx = fmaxf(mx, red[row][i]);
            float sum = 0.f;
            for (int s = sbeg; s < sbeg + T_DIM / 8; ++s) {
                float ev = __expf(sc[row][s] - mx);
                sc[row][s] = ev;
                sum += ev;
            }
            __syncthreads();           // all max-reads done before red reuse
            red[row][seg] = sum;
            __syncthreads();
            sum = 0.f;
#pragma unroll
            for (int i = 0; i < 8; ++i) sum += red[row][i];
            float inv = 1.0f / sum;
            for (int s = sbeg; s < sbeg + T_DIM / 8; ++s) {
                float p = sc[row][s] * inv;
                sc[row][s] = p;
                pf[row][s] = (half_t)p;
            }
        }
        __syncthreads();

        // ---- phase C: accumulate head-averaged weights (tile owned by this WG) ----
        {
            size_t avgbase = (size_t)b * T_DIM * T_DIM + (size_t)t0 * T_DIM;
            const float invH = 1.0f / (float)H_DIM;
            for (int idx = tid; idx < 16 * T_DIM; idx += 128) {
                int r = idx >> 11, s = idx & (T_DIM - 1);
                float p = sc[r][s] * invH;
                float* dst = avg + avgbase + (size_t)r * T_DIM + s;
                if (h == 0) *dst = p; else *dst += p;
            }
        }

        // ---- phase D: O = P V  (waves split the 4 d-tiles; A from f16 LDS) ----
        {
            int n0 = wave * 16;
            int kb = (lane >> 4) << 3;
            int row = lane & 15;
            v8f c = {};
            for (int k0 = 0; k0 < T_DIM; k0 += 32) {
                v8h lo = *(const v8h*)&pf[row][k0 + kb];
                v8h hi = *(const v8h*)&pf[row][k0 + kb + 16];
                v16h a;
#pragma unroll
                for (int i = 0; i < 8; ++i) { a[i] = lo[i]; a[8 + i] = hi[i]; }
                v16h bm = load_B32x16(V, DH, k0, n0);
                c = wmma_f16(a, bm, c);
            }
            int n = n0 + (lane & 15);
            int rbase = (lane >> 4) << 3;
#pragma unroll
            for (int r = 0; r < 8; ++r) {
                int t = t0 + rbase + r;
                attnh[((size_t)t * B_DIM + b) * E_DIM + h * DH + n] = (half_t)c[r];
            }
        }
        __syncthreads();               // sc/pf reused next head
    }
}

// ---------- GEMM2: out = attnh @ out_w^T + out_b ; 32x64 register-blocked ----------
__global__ void __launch_bounds__(32)
k_gemm_out(const half_t* __restrict__ A,     // [8192,1024] f16
           const half_t* __restrict__ Wt,    // [1024,1024] K-major f16
           const float*  __restrict__ bias,  // [1024]
           float* __restrict__ out) {        // [8192,1024] f32
    int m0 = blockIdx.x * 32;
    int n0 = blockIdx.y * 64;
    v8f acc[2][4] = {};
    for (int k0 = 0; k0 < E_DIM; k0 += 32) {
        v16h a0 = load_A16x32(A, E_DIM, m0, k0);
        v16h a1 = load_A16x32(A, E_DIM, m0 + 16, k0);
#pragma unroll
        for (int j = 0; j < 4; ++j) {
            v16h b = load_B32x16(Wt, E_DIM, k0, n0 + 16 * j);
            acc[0][j] = wmma_f16(a0, b, acc[0][j]);
            acc[1][j] = wmma_f16(a1, b, acc[1][j]);
        }
    }
    int lane = threadIdx.x & 31;
#pragma unroll
    for (int i = 0; i < 2; ++i) {
#pragma unroll
        for (int j = 0; j < 4; ++j) {
            int n = n0 + 16 * j + (lane & 15);
            float bv = bias[n];
            int mbase = m0 + 16 * i + ((lane >> 4) << 3);
#pragma unroll
            for (int r = 0; r < 8; ++r)
                out[(size_t)(mbase + r) * E_DIM + n] = acc[i][j][r] + bv;
        }
    }
}

extern "C" void kernel_launch(void* const* d_in, const int* in_sizes, int n_in,
                              void* d_out, int out_size, void* d_ws, size_t ws_size,
                              hipStream_t stream) {
    const float* x    = (const float*)d_in[0];   // [2048,4,1024]
    const float* wqkv = (const float*)d_in[1];   // [3072,1024]
    const float* bqkv = (const float*)d_in[2];   // [3072]
    const float* wout = (const float*)d_in[3];   // [1024,1024]
    const float* bout = (const float*)d_in[4];   // [1024]

    float* out_attn = (float*)d_out;                               // [2048,4,1024]
    float* out_avg  = (float*)d_out + (size_t)M_ROWS * E_DIM;      // [4,2048,2048]

    // workspace carve-up
    char* ws = (char*)d_ws;
    half_t* xb     = (half_t*)(ws);                                // 8.39M f16 (reused as attnh)
    half_t* wqkvT  = (half_t*)(ws + 16777216);                     // 3.15M f16
    half_t* woT    = (half_t*)(ws + 23068672);                     // 1.05M f16
    half_t* qh     = (half_t*)(ws + 25165824);                     // 8.39M f16
    half_t* kT     = (half_t*)(ws + 41943040);                     // 8.39M f16
    half_t* vh     = (half_t*)(ws + 58720256);                     // 8.39M f16
    half_t* attnh  = xb;                                           // reuse: x f16 dead after GEMM1

    // 1) precision conversion / weight transposes (K-major B operands)
    {
        int n = M_ROWS * E_DIM;                                    // 8,388,608
        k_cvt<<<(n + 255) / 256, 256, 0, stream>>>(x, xb, n);
    }
    {
        int n = N_QKV * E_DIM;                                     // 3,145,728
        k_tcvt<<<(n + 255) / 256, 256, 0, stream>>>(wqkv, wqkvT, N_QKV, E_DIM);
    }
    {
        int n = E_DIM * E_DIM;                                     // 1,048,576
        k_tcvt<<<(n + 255) / 256, 256, 0, stream>>>(wout, woT, E_DIM, E_DIM);
    }

    // 2) fused qkv projection -> head-major q / k^T / v (32x64 per-wave tiles)
    {
        dim3 grid(M_ROWS / 32, N_QKV / 64);                        // 256 x 48
        k_gemm_qkv<<<grid, 32, 0, stream>>>(xb, wqkvT, bqkv, qh, kT, vh);
    }

    // 3) fused attention (scores, softmax, avg weights, P@V) per (b, q-tile)
    {
        dim3 grid(T_DIM / 16, B_DIM);                              // 128 x 4
        size_t smem = (size_t)16 * T_DIM * 4                       // f32 probs
                    + (size_t)16 * T_DIM * 2                       // f16 probs
                    + 16 * 8 * sizeof(float);                      // reductions
        k_attn<<<grid, 128, smem, stream>>>(qh, kT, vh, attnh, out_avg);
    }

    // 4) output projection -> f32 d_out (32x64 per-wave tiles)
    {
        dim3 grid(M_ROWS / 32, E_DIM / 64);                        // 256 x 16
        k_gemm_out<<<grid, 32, 0, stream>>>(attnh, woT, bout, out_attn);
    }
    (void)in_sizes; (void)n_in; (void)out_size; (void)ws_size;
}
